// CustomLSTMLayer_38010460569798
// MI455X (gfx1250) — matrix operand order repaired and checked by reading, not compiled, not run on hardware
//
#include <hip/hip_runtime.h>
#include <math.h>

// Problem constants (from reference: S,B,D,H = 512,32,1024,1024)
#define SS 512
#define BB 32
#define DD 1024
#define HH 1024
#define NN (4 * HH)          // 4096 gate width
#define TC 64                // time-steps per gate-buffer chunk
#define MC (TC * BB)         // 2048 rows per chunk GEMM
#define LDSB_STRIDE 40       // shorts per B-tile row (32 data + 8 pad -> conflict-free)

typedef __attribute__((ext_vector_type(16))) __bf16 v16bf;
typedef __attribute__((ext_vector_type(8)))  __bf16 v8bf;
typedef __attribute__((ext_vector_type(8)))  float  v8f;

__device__ __forceinline__ unsigned short f2bf(float f) {
  unsigned int u = __float_as_uint(f);
  if ((u & 0x7fffffffu) > 0x7f800000u) return (unsigned short)0x7fc0; // NaN
  unsigned int r = u + 0x7fffu + ((u >> 16) & 1u);                    // RNE
  return (unsigned short)(r >> 16);
}

__device__ __forceinline__ float sigm(float x) { return 1.0f / (1.0f + expf(-x)); }

// Generic (__shared__) pointer -> 32-bit LDS byte offset.
__device__ __forceinline__ unsigned lds_offset(const void* p) {
  return (unsigned)(uintptr_t)(__attribute__((address_space(3))) const void*)p;
}

// CDNA5 async global->LDS copy, 16 bytes per lane, tracked with ASYNCcnt.
// VDST = per-lane LDS byte address, VADDR = per-lane 64-bit global address.
__device__ __forceinline__ void async_load_b128(unsigned lds_off, const void* gaddr) {
  asm volatile("global_load_async_to_lds_b128 %0, %1, off"
               :: "v"(lds_off), "v"(gaddr)
               : "memory");
}

__device__ __forceinline__ void wait_asynccnt0() {
  asm volatile("s_wait_asynccnt 0x0" ::: "memory");
}

// ---------------- setup kernels ----------------
__global__ void cvt_f32_bf16(const float* __restrict__ in,
                             unsigned short* __restrict__ out, int n) {
  int i = blockIdx.x * blockDim.x + threadIdx.x;
  if (i < n) out[i] = f2bf(in[i]);
}

__global__ void copy_f32(const float* __restrict__ in, float* __restrict__ out, int n) {
  int i = blockIdx.x * blockDim.x + threadIdx.x;
  if (i < n) out[i] = in[i];
}

// ---------------- batched input projection: G = X @ U^T + bias ----------------
// X: [rows][DD] bf16, Ub: [NN][DD] bf16, G: [rows][NN] f32.
// Block = 256 threads = 8 waves; block tile 128(M) x 64(N); wave tile 16 x 64.
// B tile staged with GLOBAL_LOAD_ASYNC_TO_LDS_B128 into double-buffered LDS
// (no VGPR round-trip; ASYNCcnt-tracked DMA overlaps the 4 WMMAs per k-step).
// A fragments double-buffered in registers.
__global__ __launch_bounds__(256)
void gemm_xu(const unsigned short* __restrict__ X,
             const unsigned short* __restrict__ Ub,
             const float* __restrict__ bih,
             const float* __restrict__ bhh,
             float* __restrict__ G) {
  __shared__ unsigned short ldsB[2][64 * LDSB_STRIDE];  // 2 x 5120 B

  const int tid  = threadIdx.x;
  const int lane = tid & 31;
  const int wave = tid >> 5;
  const int half = lane >> 4;     // wave32 lane half
  const int mr   = lane & 15;
  const int m0   = blockIdx.y * 128 + wave * 16;
  const int n0   = blockIdx.x * 64;

  // B staging assignment: 64 rows x 64 B; each lane DMAs 16 B.
  const int sn = tid >> 2;          // 0..63  (n within tile)
  const int sk = (tid & 3) * 8;     // 0,8,16,24 (k within tile)
  const unsigned short* srcB = Ub + (size_t)(n0 + sn) * DD + sk;
  const unsigned dst0 = lds_offset(&ldsB[0][sn * LDSB_STRIDE + sk]);
  const unsigned dst1 = lds_offset(&ldsB[1][sn * LDSB_STRIDE + sk]);

  const unsigned short* arow = X + (size_t)(m0 + mr) * DD;

  v8f acc[4] = {v8f{}, v8f{}, v8f{}, v8f{}};

  // Prologue: async-stage B(k=0) into buf0, load A(k=0).
  async_load_b128(dst0, srcB);
  v16bf a_cur;
  ((v8bf*)&a_cur)[0] = *(const v8bf*)(arow + 16 * half);
  ((v8bf*)&a_cur)[1] = *(const v8bf*)(arow + 8 + 16 * half);
  wait_asynccnt0();
  __syncthreads();

  for (int kb = 0; kb < DD; kb += 32) {
    const int buf  = (kb >> 5) & 1;
    const bool more = (kb + 32 < DD);
    v16bf a_nxt = {};
    if (more) {
      // DMA next B tile into the other LDS buffer; prefetch next A fragment.
      async_load_b128(buf ? dst0 : dst1, srcB + kb + 32);
      ((v8bf*)&a_nxt)[0] = *(const v8bf*)(arow + kb + 32 + 16 * half);
      ((v8bf*)&a_nxt)[1] = *(const v8bf*)(arow + kb + 40 + 16 * half);
    }
#pragma unroll
    for (int j = 0; j < 4; ++j) {
      // B fragment: lane holds col (16j+mr), K = 16*half + [0..15] of this tile.
      v16bf b = *(const v16bf*)&ldsB[buf][(16 * j + mr) * LDSB_STRIDE + 16 * half];
      acc[j] = __builtin_amdgcn_wmma_f32_16x16x32_bf16(
          false, a_cur, false, b, (short)0, acc[j], false, false);
    }
    if (more) wait_asynccnt0();   // DMA for kb+32 has landed
    __syncthreads();
    a_cur = a_nxt;
  }

#pragma unroll
  for (int j = 0; j < 4; ++j) {
    const int n = n0 + 16 * j + mr;
    const float bias = bih[n] + bhh[n];
#pragma unroll
    for (int r = 0; r < 8; ++r) {
      const int m = m0 + r + 8 * half;   // C layout: VGPR r -> row r + 8*half
      G[(size_t)m * NN + n] = acc[j][r] + bias;
    }
  }
}

// ---------------- recurrent projection: Gt += h @ V^T ----------------
// Hb: [BB][HH] bf16, Vb: [NN][HH] bf16, Gt: [BB][NN] f32 (in place).
// 512 tiles of 16x16; 64 blocks x 8 waves, one tile per wave, K = 1024.
// Both fragments double-buffered in registers so WMMA never waits on the
// loads issued in the same iteration.
__global__ __launch_bounds__(256)
void gemm_hv(const unsigned short* __restrict__ Hb,
             const unsigned short* __restrict__ Vb,
             float* __restrict__ Gt) {
  const int lane = threadIdx.x & 31;
  const int wave = threadIdx.x >> 5;
  const int tile = blockIdx.x * 8 + wave;   // 0..511
  const int m0   = (tile & 1) * 16;
  const int n0   = (tile >> 1) * 16;
  const int half = lane >> 4;
  const int mr   = lane & 15;

  const unsigned short* arow = Hb + (size_t)(m0 + mr) * HH;
  const unsigned short* bcol = Vb + (size_t)(n0 + mr) * HH;

  v8f acc = {};
  v16bf a_cur, b_cur;
  ((v8bf*)&a_cur)[0] = *(const v8bf*)(arow + 16 * half);
  ((v8bf*)&a_cur)[1] = *(const v8bf*)(arow + 8 + 16 * half);
  b_cur = *(const v16bf*)(bcol + 16 * half);

#pragma unroll 4
  for (int kb = 0; kb < HH; kb += 32) {
    v16bf a_nxt = {}, b_nxt = {};
    if (kb + 32 < HH) {
      ((v8bf*)&a_nxt)[0] = *(const v8bf*)(arow + kb + 32 + 16 * half);
      ((v8bf*)&a_nxt)[1] = *(const v8bf*)(arow + kb + 40 + 16 * half);
      b_nxt = *(const v16bf*)(bcol + kb + 32 + 16 * half);
    }
    acc = __builtin_amdgcn_wmma_f32_16x16x32_bf16(
        false, a_cur, false, b_cur, (short)0, acc, false, false);
    a_cur = a_nxt;
    b_cur = b_nxt;
  }

  const int n = n0 + mr;
#pragma unroll
  for (int r = 0; r < 8; ++r) {
    const int m = m0 + r + 8 * half;
    const size_t idx = (size_t)m * NN + n;
    Gt[idx] += acc[r];
  }
}

// ---------------- fused gate nonlinearity + state update ----------------
__global__ void lstm_gates(const float* __restrict__ Gt,  // [BB][NN]
                           float* __restrict__ c,          // [BB*HH] running cell (ws)
                           unsigned short* __restrict__ hb,// [BB*HH] bf16 next-step h (ws)
                           float* __restrict__ out, int t, int is_last) {
  const int idx = blockIdx.x * blockDim.x + threadIdx.x;
  if (idx >= BB * HH) return;
  const int b = idx >> 10;          // idx / HH
  const int j = idx & (HH - 1);     // idx % HH
  const float* g = Gt + (size_t)b * NN;

  const float i_ = sigm(g[j]);
  const float f_ = sigm(g[HH + j]);
  const float gg = tanhf(g[2 * HH + j]);
  const float o_ = sigm(g[3 * HH + j]);

  const float cn = f_ * c[idx] + i_ * gg;
  const float hn = o_ * tanhf(cn);

  c[idx]  = cn;
  hb[idx] = f2bf(hn);

  // out tuple: (all_h, h_f, c_f, all_h, all_c) concatenated flat
  const size_t OFF1 = (size_t)SS * BB * HH;
  const size_t OFF2 = OFF1 + (size_t)BB * HH;
  const size_t OFF3 = OFF2 + (size_t)BB * HH;
  const size_t OFF4 = OFF3 + (size_t)SS * BB * HH;
  const size_t base = (size_t)t * (BB * HH);

  out[base + idx]        = hn;   // all_h (copy 1)
  out[OFF3 + base + idx] = hn;   // all_h (copy 2)
  out[OFF4 + base + idx] = cn;   // all_c
  if (is_last) {
    out[OFF1 + idx] = hn;        // h_f
    out[OFF2 + idx] = cn;        // c_f
  }
}

// ---------------- host launcher ----------------
extern "C" void kernel_launch(void* const* d_in, const int* in_sizes, int n_in,
                              void* d_out, int out_size, void* d_ws, size_t ws_size,
                              hipStream_t stream) {
  (void)in_sizes; (void)n_in; (void)out_size; (void)ws_size;
  const float* x   = (const float*)d_in[0];   // [SS][BB][DD]
  const float* h0  = (const float*)d_in[1];   // [BB][HH]
  const float* c0  = (const float*)d_in[2];   // [BB][HH]
  const float* U   = (const float*)d_in[3];   // [NN][DD]
  const float* V   = (const float*)d_in[4];   // [NN][HH]
  const float* bih = (const float*)d_in[5];   // [NN]
  const float* bhh = (const float*)d_in[6];   // [NN]
  float* out = (float*)d_out;

  // Workspace layout (~84 MB):
  //   G  : MC*NN f32  gate chunk (64 steps)        33.6 MB
  //   Xb : SS*BB*DD bf16                           33.6 MB
  //   Ub : NN*DD bf16                               8.4 MB
  //   Vb : NN*HH bf16                               8.4 MB
  //   Hb : BB*HH bf16                              64 KB
  //   Cw : BB*HH f32                              128 KB
  char* ws = (char*)d_ws;
  float*          G  = (float*)ws;
  unsigned short* Xb = (unsigned short*)(ws + (size_t)MC * NN * 4);
  unsigned short* Ub = Xb + (size_t)SS * BB * DD;
  unsigned short* Vb = Ub + (size_t)NN * DD;
  unsigned short* Hb = Vb + (size_t)NN * HH;
  float*          Cw = (float*)(Hb + (size_t)BB * HH);

  const int nx = SS * BB * DD;   // 16,777,216
  const int nu = NN * DD;        //  4,194,304
  const int nv = NN * HH;        //  4,194,304
  const int ns = BB * HH;        //     32,768

  cvt_f32_bf16<<<(nx + 255) / 256, 256, 0, stream>>>(x,  Xb, nx);
  cvt_f32_bf16<<<(nu + 255) / 256, 256, 0, stream>>>(U,  Ub, nu);
  cvt_f32_bf16<<<(nv + 255) / 256, 256, 0, stream>>>(V,  Vb, nv);
  cvt_f32_bf16<<<(ns + 255) / 256, 256, 0, stream>>>(h0, Hb, ns);
  copy_f32    <<<(ns + 255) / 256, 256, 0, stream>>>(c0, Cw, ns);

  const dim3 gdim(NN / 64, MC / 128);   // 64 x 16 blocks per chunk GEMM
  for (int tc = 0; tc < SS; tc += TC) {
    // Batched input projection for this 64-step chunk.
    gemm_xu<<<gdim, 256, 0, stream>>>(Xb + (size_t)tc * BB * DD, Ub, bih, bhh, G);
    // Sequential recurrence within the chunk.
    for (int t = tc; t < tc + TC; ++t) {
      float* Gt = G + (size_t)(t - tc) * BB * NN;
      gemm_hv   <<<64, 256, 0, stream>>>(Hb, Vb, Gt);
      lstm_gates<<<(ns + 255) / 256, 256, 0, stream>>>(Gt, Cw, Hb, out, t, t == SS - 1 ? 1 : 0);
    }
  }
}
